// MemoryAwareAKTAttention_23751169147110
// MI455X (gfx1250) — matrix operand order repaired and checked by reading, not compiled, not run on hardware
//
#include <hip/hip_runtime.h>
#include <hip/hip_bf16.h>

// ---------------- problem constants ----------------
// B=8, S=1024, E=512, H=8, D=64, BH=64
#define BB 8
#define SS 1024
#define EE 512
#define HH 8
#define DD 64
#define BH 64

typedef __attribute__((ext_vector_type(16))) _Float16 v16h;
typedef __attribute__((ext_vector_type(8)))  float    v8f;

union F16Frag { uint4 u[2]; v16h v; };

__device__ __forceinline__ v8f zero8() {
  v8f z = {0.f,0.f,0.f,0.f,0.f,0.f,0.f,0.f};
  return z;
}

__device__ __forceinline__ v8f wmma16(v16h a, v16h b, v8f c) {
  return __builtin_amdgcn_wmma_f32_16x16x32_f16(
      /*neg_a=*/false, a, /*neg_b=*/false, b,
      /*c_mod=*/(short)0, c, /*reuse_a=*/false, /*reuse_b=*/false);
}

// A fragment: 16(M) x 32(K), 16-bit. Lane l: m=l&15, h=l>>4.
// elems 0..7  <- k = 8h + 0..7 ;  elems 8..15 <- k = 16 + 8h + 0..7
__device__ __forceinline__ v16h load_frag_a(const _Float16* base, int ld, int lane) {
  int m = lane & 15, h = lane >> 4;
  const _Float16* p = base + (size_t)m * ld + 8 * h;
  F16Frag f;
  f.u[0] = *reinterpret_cast<const uint4*>(p);
  f.u[1] = *reinterpret_cast<const uint4*>(p + 16);
  return f.v;
}

// B fragment: 32(K) x 16(N), built from row-major W[N,K] (torch-style weight).
// Lane l: n=l&15, h=l>>4; elems 0..7 <- k = 8h + 0..7 ; elems 8..15 <- k = 16 + 8h + 0..7
__device__ __forceinline__ v16h load_frag_b(const _Float16* base, int ld, int lane) {
  int n = lane & 15, h = lane >> 4;
  const _Float16* p = base + (size_t)n * ld + 8 * h;
  F16Frag f;
  f.u[0] = *reinterpret_cast<const uint4*>(p);
  f.u[1] = *reinterpret_cast<const uint4*>(p + 16);
  return f.v;
}

// ---------------- elementwise converts ----------------
__global__ void cvt_f32_f16(const float* __restrict__ src, _Float16* __restrict__ dst, int n) {
  int i = blockIdx.x * blockDim.x + threadIdx.x;
  if (i < n) dst[i] = (_Float16)src[i];
}

// vT layout: [BH][80][S]; rows 64..79: row64 = ones (rowsum trick), rest zeros
__global__ void vT_pad_init(_Float16* __restrict__ vT) {
  int i = blockIdx.x * blockDim.x + threadIdx.x; // 64*16*1024
  int t  = i & 1023;
  int c  = (i >> 10) & 15;
  int bh = i >> 14;
  vT[(size_t)bh * 80 * SS + (size_t)(64 + c) * SS + t] = (c == 0) ? (_Float16)1.0f : (_Float16)0.0f;
}

// Software-pipelined 16xK x Kx64 GEMM body: loads for chunk k+32 are issued
// before the WMMAs of chunk k (both generations live), so the fetches stream
// under the matrix ops instead of serializing on s_wait_loadcnt 0.
__device__ __forceinline__ void gemm16x64_pipe(const _Float16* arow, int lda,
                                               const _Float16* wbase, int ldw,
                                               int K, int lane, v8f acc[4]) {
  v16h a  = load_frag_a(arow, lda, lane);
  v16h b0 = load_frag_b(wbase + (size_t)0  * ldw, ldw, lane);
  v16h b1 = load_frag_b(wbase + (size_t)16 * ldw, ldw, lane);
  v16h b2 = load_frag_b(wbase + (size_t)32 * ldw, ldw, lane);
  v16h b3 = load_frag_b(wbase + (size_t)48 * ldw, ldw, lane);
  #pragma unroll 4
  for (int kc = 32; kc < K; kc += 32) {
    v16h an = load_frag_a(arow + kc, lda, lane);
    v16h c0 = load_frag_b(wbase + (size_t)0  * ldw + kc, ldw, lane);
    v16h c1 = load_frag_b(wbase + (size_t)16 * ldw + kc, ldw, lane);
    v16h c2 = load_frag_b(wbase + (size_t)32 * ldw + kc, ldw, lane);
    v16h c3 = load_frag_b(wbase + (size_t)48 * ldw + kc, ldw, lane);
    acc[0] = wmma16(a, b0, acc[0]);
    acc[1] = wmma16(a, b1, acc[1]);
    acc[2] = wmma16(a, b2, acc[2]);
    acc[3] = wmma16(a, b3, acc[3]);
    a = an; b0 = c0; b1 = c1; b2 = c2; b3 = c3;
  }
  acc[0] = wmma16(a, b0, acc[0]);
  acc[1] = wmma16(a, b1, acc[1]);
  acc[2] = wmma16(a, b2, acc[2]);
  acc[3] = wmma16(a, b3, acc[3]);
}

// ---------------- GEMM1: integrated = [x|mem] @ Wm^T + bm ----------------
// M=8192, N=512, K=1024. Wave computes 16x64. Output f16.
__global__ void gemm_integrate(const _Float16* __restrict__ xf, const _Float16* __restrict__ mf,
                               const _Float16* __restrict__ wm, const float* __restrict__ bm,
                               _Float16* __restrict__ integ) {
  int lane = threadIdx.x & 31, wid = threadIdx.x >> 5;
  int wg = blockIdx.x * (blockDim.x >> 5) + wid;
  int row0 = (wg >> 3) * 16;
  int n0   = (wg & 7) * 64;
  v8f acc[4] = {zero8(), zero8(), zero8(), zero8()};
  const _Float16* xrow = xf + (size_t)row0 * EE;
  const _Float16* mrow = mf + (size_t)row0 * EE;
  const _Float16* wbase = wm + (size_t)n0 * (2 * EE);
  gemm16x64_pipe(xrow, EE, wbase,      2 * EE, EE, lane, acc);  // k = 0..511 (x)
  gemm16x64_pipe(mrow, EE, wbase + EE, 2 * EE, EE, lane, acc);  // k = 512..1023 (mem)
  int n = lane & 15, h = lane >> 4;
  for (int nb = 0; nb < 4; nb++) {
    int col = n0 + nb * 16 + n;
    float bias = bm[col];
    for (int r = 0; r < 8; r++) {
      int m = r + 8 * h;
      integ[(size_t)(row0 + m) * EE + col] = (_Float16)(acc[nb][r] + bias);
    }
  }
}

// ---------------- GEMM2: q/k/v projections ----------------
// blockIdx.y selects 0=Q (scaled by 1/sqrt(D)), 1=K, 2=V(transposed+pad layout)
__global__ void gemm_qkv(const _Float16* __restrict__ integ,
                         const _Float16* __restrict__ wq, const _Float16* __restrict__ wk,
                         const _Float16* __restrict__ wv,
                         _Float16* __restrict__ q16, _Float16* __restrict__ k16,
                         _Float16* __restrict__ vT) {
  int which = blockIdx.y;
  const _Float16* w = (which == 0) ? wq : (which == 1) ? wk : wv;
  int lane = threadIdx.x & 31, wid = threadIdx.x >> 5;
  int wg = blockIdx.x * (blockDim.x >> 5) + wid;
  int row0 = (wg >> 3) * 16;
  int n0   = (wg & 7) * 64;
  v8f acc[4] = {zero8(), zero8(), zero8(), zero8()};
  gemm16x64_pipe(integ + (size_t)row0 * EE, EE, w + (size_t)n0 * EE, EE, EE, lane, acc);
  int n = lane & 15, h = lane >> 4;
  int bb = row0 >> 10;           // batch
  int sbase = row0 & 1023;       // position within batch
  for (int nb = 0; nb < 4; nb++) {
    int col = n0 + nb * 16 + n;
    int hd = col >> 6, d = col & 63;
    for (int r = 0; r < 8; r++) {
      int s = sbase + r + 8 * h;
      float val = acc[nb][r];
      size_t bh = (size_t)bb * HH + hd;
      if (which == 0)
        q16[(bh * SS + s) * DD + d] = (_Float16)(val * 0.125f);   // fold 1/sqrt(64)
      else if (which == 1)
        k16[(bh * SS + s) * DD + d] = (_Float16)val;
      else
        vT[bh * 80 * SS + (size_t)d * SS + s] = (_Float16)val;
    }
  }
}

// ---------------- fused decayed attention ----------------
// Wave: 16 query rows x all 1024 keys. Writes unnormalized p to attn,
// accumulates out = sum(p*v) and rowsum via ones-column of vT.
__global__ void attn_fused(const _Float16* __restrict__ q16, const _Float16* __restrict__ k16,
                           const _Float16* __restrict__ vT, const float* __restrict__ gammas,
                           float* __restrict__ attn, float* __restrict__ rsbuf,
                           _Float16* __restrict__ oh16) {
  __shared__ __align__(16) _Float16 lds[8][16 * 32];
  int lane = threadIdx.x & 31, wid = threadIdx.x >> 5;
  int wg = blockIdx.x * (blockDim.x >> 5) + wid;
  int bh = wg >> 6;            // 0..63
  int s0 = (wg & 63) * 16;     // query tile
  int hidx = bh & 7;
  float g = gammas[hidx];
  float gamma = log1pf(__expf(g));   // softplus

  const _Float16* qrow = q16 + ((size_t)bh * SS + s0) * DD;
  v16h qa0 = load_frag_a(qrow, DD, lane);        // K 0..31
  v16h qa1 = load_frag_a(qrow + 32, DD, lane);   // K 32..63

  v8f outc[5] = {zero8(), zero8(), zero8(), zero8(), zero8()};
  int n = lane & 15, hh = lane >> 4;
  float* attnbh = attn + (size_t)bh * SS * SS;
  _Float16* pl = &lds[wid][0];
  const _Float16* vbh = vT + (size_t)bh * 80 * SS;
  const _Float16* kbh = k16 + (size_t)bh * SS * DD;

  for (int t0 = 0; t0 < SS; t0 += 32) {
    // prefetch v fragments for this chunk early: consumed only after the
    // score WMMAs + elementwise exp work below
    v16h vb0 = load_frag_b(vbh + (size_t)0  * SS + t0, SS, lane);
    v16h vb1 = load_frag_b(vbh + (size_t)16 * SS + t0, SS, lane);
    v16h vb2 = load_frag_b(vbh + (size_t)32 * SS + t0, SS, lane);
    v16h vb3 = load_frag_b(vbh + (size_t)48 * SS + t0, SS, lane);
    v16h vb4 = load_frag_b(vbh + (size_t)64 * SS + t0, SS, lane);  // ones column -> rowsum
    #pragma unroll
    for (int st = 0; st < 2; st++) {
      int c0 = t0 + st * 16;
      const _Float16* kbase = kbh + (size_t)c0 * DD;
      v16h kb0 = load_frag_b(kbase, DD, lane);
      v16h kb1 = load_frag_b(kbase + 32, DD, lane);
      v8f sc = zero8();
      sc = wmma16(qa0, kb0, sc);
      sc = wmma16(qa1, kb1, sc);
      int j = c0 + n;
      #pragma unroll
      for (int r = 0; r < 8; r++) {
        int i = s0 + r + 8 * hh;
        float dist = fabsf((float)(i - j));
        float decay = __expf(-gamma * dist);
        float p = __expf(sc[r] * decay);
        p = fminf(fmaxf(p, 1e-5f), 1e5f);
        attnbh[(size_t)i * SS + j] = p;              // unnormalized; pass 2 divides
        pl[(r + 8 * hh) * 32 + st * 16 + n] = (_Float16)p;  // per-wave LDS, DS ops in-order
      }
    }
    v16h pa = load_frag_a(pl, 32, lane);             // p tile as A operand
    outc[0] = wmma16(pa, vb0, outc[0]);
    outc[1] = wmma16(pa, vb1, outc[1]);
    outc[2] = wmma16(pa, vb2, outc[2]);
    outc[3] = wmma16(pa, vb3, outc[3]);
    outc[4] = wmma16(pa, vb4, outc[4]);
  }

  for (int r = 0; r < 8; r++) {
    float rs = __shfl(outc[4][r], hh * 16, 32);      // col 64 lives at n==0 lanes
    float inv = 1.0f / (rs + 1e-9f);
    int s = s0 + r + 8 * hh;
    if (n == 0) rsbuf[(size_t)bh * SS + s] = rs;
    for (int nb = 0; nb < 4; nb++) {
      int d = nb * 16 + n;
      oh16[((size_t)bh * SS + s) * DD + d] = (_Float16)(outc[nb][r] * inv);
    }
  }
}

// ---------------- attn normalization pass ----------------
__global__ void normalize_attn(float* __restrict__ attn, const float* __restrict__ rsbuf) {
  size_t idx = (size_t)blockIdx.x * blockDim.x + threadIdx.x;  // one float4 each
  size_t e0 = idx * 4;
  float inv = 1.0f / (rsbuf[e0 >> 10] + 1e-9f);
  float4* a4 = reinterpret_cast<float4*>(attn) + idx;
  float4 v = *a4;
  v.x *= inv; v.y *= inv; v.z *= inv; v.w *= inv;
  *a4 = v;
}

// ---------------- output projection: out = OH(gathered) @ Wo^T + bo ----------------
__global__ void gemm_out(const _Float16* __restrict__ oh16, const _Float16* __restrict__ wo,
                         const float* __restrict__ bo, float* __restrict__ out) {
  int lane = threadIdx.x & 31, wid = threadIdx.x >> 5;
  int wg = blockIdx.x * (blockDim.x >> 5) + wid;
  int row0 = (wg >> 3) * 16;
  int n0   = (wg & 7) * 64;
  int bb = row0 >> 10, sbase = row0 & 1023;
  v8f acc[4] = {zero8(), zero8(), zero8(), zero8()};
  const _Float16* abh   = oh16 + ((size_t)bb * HH * SS + sbase) * DD;  // + hd*SS*DD per head
  const _Float16* wbase = wo + (size_t)n0 * EE;
  // software-pipelined over K with per-head A gather
  v16h a  = load_frag_a(abh, DD, lane);
  v16h b0 = load_frag_b(wbase + (size_t)0  * EE, EE, lane);
  v16h b1 = load_frag_b(wbase + (size_t)16 * EE, EE, lane);
  v16h b2 = load_frag_b(wbase + (size_t)32 * EE, EE, lane);
  v16h b3 = load_frag_b(wbase + (size_t)48 * EE, EE, lane);
  #pragma unroll 4
  for (int kc = 32; kc < EE; kc += 32) {
    int hd = kc >> 6;
    v16h an = load_frag_a(abh + (size_t)hd * SS * DD + (kc & 32), DD, lane);
    v16h c0 = load_frag_b(wbase + (size_t)0  * EE + kc, EE, lane);
    v16h c1 = load_frag_b(wbase + (size_t)16 * EE + kc, EE, lane);
    v16h c2 = load_frag_b(wbase + (size_t)32 * EE + kc, EE, lane);
    v16h c3 = load_frag_b(wbase + (size_t)48 * EE + kc, EE, lane);
    acc[0] = wmma16(a, b0, acc[0]);
    acc[1] = wmma16(a, b1, acc[1]);
    acc[2] = wmma16(a, b2, acc[2]);
    acc[3] = wmma16(a, b3, acc[3]);
    a = an; b0 = c0; b1 = c1; b2 = c2; b3 = c3;
  }
  acc[0] = wmma16(a, b0, acc[0]);
  acc[1] = wmma16(a, b1, acc[1]);
  acc[2] = wmma16(a, b2, acc[2]);
  acc[3] = wmma16(a, b3, acc[3]);
  int n = lane & 15, h = lane >> 4;
  for (int nb = 0; nb < 4; nb++) {
    int col = n0 + nb * 16 + n;
    float bias = bo[col];
    for (int r = 0; r < 8; r++) {
      int m = r + 8 * h;
      out[(size_t)(row0 + m) * EE + col] = acc[nb][r] + bias;
    }
  }
}

// ---------------- launch ----------------
extern "C" void kernel_launch(void* const* d_in, const int* in_sizes, int n_in,
                              void* d_out, int out_size, void* d_ws, size_t ws_size,
                              hipStream_t stream) {
  (void)in_sizes; (void)n_in; (void)out_size; (void)ws_size;
  const float* x    = (const float*)d_in[0];
  const float* mem  = (const float*)d_in[1];
  const float* Wq   = (const float*)d_in[2];
  const float* Wk   = (const float*)d_in[3];
  const float* Wv   = (const float*)d_in[4];
  const float* Wm   = (const float*)d_in[5];
  const float* bm   = (const float*)d_in[6];
  const float* Wo   = (const float*)d_in[7];
  const float* bo   = (const float*)d_in[8];
  const float* gam  = (const float*)d_in[9];

  float* out  = (float*)d_out;                         // [8,1024,512]
  float* attn = (float*)d_out + (size_t)BB * SS * EE;  // [8,8,1024,1024]

  // workspace layout (halfs)
  constexpr size_t NX   = (size_t)BB * SS * EE;        // 4194304
  constexpr size_t OFF_X   = 0;
  constexpr size_t OFF_MEM = OFF_X + NX;
  constexpr size_t OFF_WM  = OFF_MEM + NX;
  constexpr size_t OFF_WQ  = OFF_WM + (size_t)EE * 2 * EE;
  constexpr size_t OFF_WK  = OFF_WQ + (size_t)EE * EE;
  constexpr size_t OFF_WV  = OFF_WK + (size_t)EE * EE;
  constexpr size_t OFF_WO  = OFF_WV + (size_t)EE * EE;
  constexpr size_t OFF_INT = OFF_WO + (size_t)EE * EE;
  constexpr size_t OFF_Q   = OFF_INT + NX;
  constexpr size_t OFF_K   = OFF_Q + (size_t)BH * SS * DD;
  constexpr size_t OFF_VT  = OFF_K + (size_t)BH * SS * DD;
  constexpr size_t OFF_OH  = OFF_VT + (size_t)BH * 80 * SS;
  constexpr size_t HALF_TOTAL = OFF_OH + (size_t)BH * SS * DD;

  _Float16* h = (_Float16*)d_ws;
  _Float16 *xf = h + OFF_X, *mf = h + OFF_MEM, *wm16 = h + OFF_WM;
  _Float16 *wq16 = h + OFF_WQ, *wk16 = h + OFF_WK, *wv16 = h + OFF_WV, *wo16 = h + OFF_WO;
  _Float16 *integ = h + OFF_INT, *q16 = h + OFF_Q, *k16 = h + OFF_K;
  _Float16 *vT = h + OFF_VT, *oh16 = h + OFF_OH;
  float* rsbuf = (float*)((char*)d_ws + HALF_TOTAL * sizeof(_Float16));

  auto cvt = [&](const float* s, _Float16* d, int nel) {
    cvt_f32_f16<<<(nel + 255) / 256, 256, 0, stream>>>(s, d, nel);
  };
  cvt(x,   xf,   (int)NX);
  cvt(mem, mf,   (int)NX);
  cvt(Wm,  wm16, EE * 2 * EE);
  cvt(Wq,  wq16, EE * EE);
  cvt(Wk,  wk16, EE * EE);
  cvt(Wv,  wv16, EE * EE);
  cvt(Wo,  wo16, EE * EE);
  vT_pad_init<<<(BH * 16 * SS) / 256, 256, 0, stream>>>(vT);

  // 512 M-tiles x 8 N-groups = 4096 waves -> 512 blocks of 8 waves
  gemm_integrate<<<512, 256, 0, stream>>>(xf, mf, wm16, bm, integ);
  gemm_qkv<<<dim3(512, 3), 256, 0, stream>>>(integ, wq16, wk16, wv16, q16, k16, vT);
  attn_fused<<<512, 256, 0, stream>>>(q16, k16, vT, gam, attn, rsbuf, oh16);
  normalize_attn<<<(BH * SS * SS / 4) / 256, 256, 0, stream>>>(attn, rsbuf);
  gemm_out<<<512, 256, 0, stream>>>(oh16, wo16, bo, out);
}
